// KernelAttention_21088289423968
// MI455X (gfx1250) — matrix operand order repaired and checked by reading, not compiled, not run on hardware
//
#include <hip/hip_runtime.h>
#include <hip/hip_bf16.h>
#include <math.h>

// ---------------------------------------------------------------------------
// Performer-style random-feature attention for MI455X (gfx1250, wave32).
// GEMMs: v_wmma_f32_16x16x32_bf16. z staging in the reduction pass uses the
// Tensor Data Mover (tensor_load_to_lds, double-buffered, s_wait_tensorcnt).
// B operands live in LDS pre-swizzled to fragment order -> ds_load_b128.
// ---------------------------------------------------------------------------

namespace {

constexpr int   kN     = 100000;               // nodes
constexpr int   kH     = 8;                    // heads
constexpr int   kDH    = 32;                   // head dim
constexpr int   kM     = 64;                   // random features
constexpr int   kNT    = 64;                   // node tile (rows per block)
constexpr int   kS     = 64;                   // kvs reduction slices per head
constexpr int   kTiles = (kN + kNT - 1) / kNT;
constexpr int   kDDld  = 68;                   // padded stride for dd tile
constexpr float kEps   = 1e-4f;
constexpr float kRatio = 0.125f;               // M^-0.5
constexpr float kDN    = 0.42044820762685725f; // DH^-0.25

typedef __attribute__((ext_vector_type(16))) __bf16 v16bf;
typedef __attribute__((ext_vector_type(8)))  float  v8f;
typedef __attribute__((ext_vector_type(4)))  unsigned int u32x4;
typedef __attribute__((ext_vector_type(8)))  int    i32x8;
typedef __attribute__((ext_vector_type(4)))  int    i32x4;

// --- WMMA fragment helpers (wave32; C/D per ISA 7.12.2; A/B use contiguous k = 16*(lane/16)+s) ----------
// A (16x32 bf16), row-major source: lane holds one contiguous 16-half run.
__device__ inline v16bf lds_load_a(const __bf16* s, int ld, int row0) {
  int lane = threadIdx.x & 31;
  int row  = row0 + (lane & 15);
  int kb   = (lane & 16) ? 16 : 0;
  const __bf16* base = s + row * ld + kb;
  union { uint4 u[2]; v16bf v; } c;
  c.u[0] = *(const uint4*)(base);
  c.u[1] = *(const uint4*)(base + 8);
  return c.v;
}

// B operand kept pre-swizzled in LDS: [tile][lane][16 halves] contiguous.
// Element (k, n) of a (K<=64 x N) panel with nc col-tiles:
__device__ inline int bfrag_idx64(int k, int n, int nc) {
  int tile = (k >> 5) * nc + (n >> 4);
  int lane = (n & 15) | (k & 16);
  return ((tile << 5) + lane) * 16 + (k & 15);
}
__device__ inline v16bf lds_load_bfrag(const __bf16* s, int tile) {
  int lane = threadIdx.x & 31;
  const uint4* p = (const uint4*)(s + (((tile << 5) + lane) << 4));
  union { uint4 u[2]; v16bf v; } c;
  c.u[0] = p[0];
  c.u[1] = p[1];
  return c.v;
}

// C/D (16x16 f32): VGPR i, lane l -> row = i + 8*(l/16), col = l%16.
__device__ inline void lds_store_c(float* s, int ld, int row0, int col0, v8f c) {
  int lane = threadIdx.x & 31;
  int col  = col0 + (lane & 15);
  int rb   = row0 + ((lane & 16) ? 8 : 0);
#pragma unroll
  for (int i = 0; i < 8; ++i) s[(rb + i) * ld + col] = c[i];
}

__device__ inline v8f wmma_bf16(v16bf a, v16bf b, v8f c) {
  return __builtin_amdgcn_wmma_f32_16x16x32_bf16(
      false, a, false, b, (short)0, c, false, false);
}

// monotone f32 <-> u32 encoding so atomicMax(u32) gives float max
__device__ inline unsigned enc_f32(float x) {
  unsigned u = __float_as_uint(x);
  return (u & 0x80000000u) ? ~u : (u | 0x80000000u);
}
__device__ inline float dec_f32(unsigned u) {
  unsigned b = (u & 0x80000000u) ? (u ^ 0x80000000u) : ~u;
  return __uint_as_float(b);
}

// --- Tensor Data Mover: 2-D tile load (rows x 32 f32, row stride 256) ------
// D# per CDNA5 ISA ch.8: group0 = {flags, lds_addr, global_addr, type=2},
// group1 = {data_size, dims/strides, tile dims}. Groups 2/3 unused (2-D).
// This toolchain exposes the 6-arg builtin (g0,g1,g2,g3,g4,cpol).
__device__ inline void tdm_load_tile(const float* gsrc, unsigned lds_off,
                                     unsigned rows_avail) {
  unsigned long long ga = (unsigned long long)(uintptr_t)gsrc;
  u32x4 g0;
  g0[0] = 1u;                                       // count=1 (user D#)
  g0[1] = lds_off;                                  // lds_addr (bytes)
  g0[2] = (unsigned)(ga & 0xffffffffull);           // global_addr[31:0]
  g0[3] = (unsigned)((ga >> 32) & 0x01ffffffull) | (2u << 30);  // type=2
  i32x8 g1;
  g1[0] = (int)(2u << 16);                          // data_size = 4 bytes
  g1[1] = (int)((256u & 0xffffu) << 16);            // tensor_dim0[15:0]
  g1[2] = (int)((256u >> 16) | ((rows_avail & 0xffffu) << 16));  // dim1 lo
  g1[3] = (int)((rows_avail >> 16) | (32u << 16));  // dim1 hi | tile_dim0=32
  g1[4] = (int)(unsigned)kNT;                       // tile_dim1 = 64
  g1[5] = (int)256u;                                // dim0_stride[31:0]
  g1[6] = 0;                                        // dim0_stride hi | s1 lo
  g1[7] = 0;
  i32x4 zz4 = {0, 0, 0, 0};
  i32x8 zz8 = {0, 0, 0, 0, 0, 0, 0, 0};
  __builtin_amdgcn_tensor_load_to_lds(g0, g1, zz4, zz4, zz8, 0);
}

// ---------------------------------------------------------------------------
__global__ __launch_bounds__(256) void init_kmax(unsigned* kmax_bits) {
  if (threadIdx.x < kH) kmax_bits[threadIdx.x] = 0u;  // encodes very negative
}

// ---------------------------------------------------------------------------
// Pass A: per-head global max of dd_k = (s * (z@Wk + b)) . proj^T
__global__ __launch_bounds__(256) void kmax_kernel(
    const float* __restrict__ z, const float* __restrict__ tau,
    const float* __restrict__ Wk_w, const float* __restrict__ Wk_b,
    const float* __restrict__ proj, unsigned* __restrict__ kmax_bits) {
  __shared__ __bf16 sZ[kNT * 40] __attribute__((aligned(16)));
  __shared__ __bf16 sW[1024]     __attribute__((aligned(16)));  // B-frag
  __shared__ __bf16 sPT[2048]    __attribute__((aligned(16)));  // B-frag
  __shared__ __bf16 sK[kNT * 40] __attribute__((aligned(16)));
  __shared__ float  sDD[kNT * kDDld];
  __shared__ float  sRed[256];

  const int   t  = threadIdx.x;
  const int   w  = t >> 5;
  const int   h  = blockIdx.y;
  const int   n0 = blockIdx.x * kNT;
  const float sc = rsqrtf(tau[0]) * kDN;

  for (int i = t; i < 32 * 32; i += 256)
    sW[bfrag_idx64(i >> 5, i & 31, 2)] = (__bf16)Wk_w[h * 32 * 32 + i];
  for (int i = t; i < kM * 32; i += 256)
    sPT[bfrag_idx64(i & 31, i >> 5, 4)] = (__bf16)proj[i];
  {
    int row = t >> 2, c0 = (t & 3) * 8;
    int n = n0 + row;
    const float* src = z + (size_t)n * (kH * kDH) + h * kDH + c0;
    union { __bf16 hx[8]; uint4 u; } pk;
#pragma unroll
    for (int j = 0; j < 8; ++j)
      pk.hx[j] = (__bf16)((n < kN) ? src[j] : 0.0f);
    *(uint4*)&sZ[row * 40 + c0] = pk.u;
  }
  __syncthreads();

  {  // GEMM1: k' = sc * (z @ Wk + b)   (one 16x16 tile per wave)
    v8f acc = {};
    acc = wmma_bf16(lds_load_a(sZ, 40, (w >> 1) * 16),
                    lds_load_bfrag(sW, w & 1), acc);
    int lane = t & 31;
    int col  = (w & 1) * 16 + (lane & 15);
    int rb   = (w >> 1) * 16 + ((lane & 16) ? 8 : 0);
    float bias = Wk_b[h * kDH + col];
#pragma unroll
    for (int i = 0; i < 8; ++i)
      sK[(rb + i) * 40 + col] = (__bf16)(sc * (acc[i] + bias));
  }
  __syncthreads();

#pragma unroll
  for (int i = 0; i < 2; ++i) {  // GEMM2: dd = k' @ proj^T (two tiles/wave)
    int tid = w * 2 + i;
    int r = tid >> 2, c = tid & 3;
    v8f acc = {};
    acc = wmma_bf16(lds_load_a(sK, 40, r * 16), lds_load_bfrag(sPT, c), acc);
    lds_store_c(sDD, kDDld, r * 16, c * 16, acc);
  }
  __syncthreads();

  float mx = -3.0e38f;
  {
    int row = t >> 2, c0 = (t & 3) * 16;
    if (n0 + row < kN) {
#pragma unroll
      for (int j = 0; j < 16; ++j) mx = fmaxf(mx, sDD[row * kDDld + c0 + j]);
    }
  }
  sRed[t] = mx;
  __syncthreads();
  for (int off = 128; off > 0; off >>= 1) {
    if (t < off) sRed[t] = fmaxf(sRed[t], sRed[t + off]);
    __syncthreads();
  }
  if (t == 0) atomicMax(&kmax_bits[h], enc_f32(sRed[0]));
}

// ---------------------------------------------------------------------------
// Pass B: per-slice partial kvs[h,m,d] = sum_n kk[n,m]*v[n,d] and ksum[h,m].
// z tiles arrive via TDM (double-buffered); WMMA C-fragments persist across
// the tile loop (K reduction over nodes). Deterministic (no float atomics).
__global__ __launch_bounds__(256) void kvs_kernel(
    const float* __restrict__ z, const float* __restrict__ tau,
    const float* __restrict__ Wk_w, const float* __restrict__ Wk_b,
    const float* __restrict__ Wv_w, const float* __restrict__ Wv_b,
    const float* __restrict__ proj, const unsigned* __restrict__ kmax_bits,
    float* __restrict__ pkvs, float* __restrict__ pksum) {
  __shared__ float  sZf[2][kNT * kDH] __attribute__((aligned(16)));  // TDM dst
  __shared__ __bf16 sZ[kNT * 40]   __attribute__((aligned(16)));
  __shared__ __bf16 sWk[1024]      __attribute__((aligned(16)));  // B-frag
  __shared__ __bf16 sWv[1024]      __attribute__((aligned(16)));  // B-frag
  __shared__ __bf16 sPT[2048]      __attribute__((aligned(16)));  // B-frag
  __shared__ __bf16 sK[kNT * 40]   __attribute__((aligned(16)));
  __shared__ __bf16 sVf[2048]      __attribute__((aligned(16)));  // B-frag K=64
  __shared__ __bf16 sKKT[kM * 72]  __attribute__((aligned(16))); // [m][row]
  __shared__ float  sDD[kNT * kDDld];
  __shared__ float  sDiag[kNT];

  const int   t     = threadIdx.x;
  const int   w     = t >> 5;
  const int   h     = blockIdx.y;
  const int   slice = blockIdx.x;
  const float sc    = rsqrtf(tau[0]) * kDN;
  const float kmax  = dec_f32(kmax_bits[h]);

  for (int i = t; i < 32 * 32; i += 256) {
    sWk[bfrag_idx64(i >> 5, i & 31, 2)] = (__bf16)Wk_w[h * 32 * 32 + i];
    sWv[bfrag_idx64(i >> 5, i & 31, 2)] = (__bf16)Wv_w[h * 32 * 32 + i];
  }
  for (int i = t; i < kM * 32; i += 256)
    sPT[bfrag_idx64(i & 31, i >> 5, 4)] = (__bf16)proj[i];

  // prologue: DMA the first tile into buffer 0
  if (w == 0 && slice < kTiles)
    tdm_load_tile(z + (size_t)slice * kNT * (kH * kDH) + h * kDH,
                  (unsigned)(uintptr_t)(const void*)sZf[0],
                  (unsigned)(kN - slice * kNT));

  v8f   kvacc = {};   // this wave's 16x16 tile of kvs (persistent accumulator)
  float ksacc = 0.0f; // ksum for m = t (t < 64)
  int   cur   = 0;

  for (int tile = slice; tile < kTiles; tile += kS, cur ^= 1) {
    const int n0 = tile * kNT;
    __syncthreads();  // previous iteration done with sZ/sK/sVf/sKKT/sZf[cur^1]
    if (w == 0) {
      int nxt = tile + kS;
      if (nxt < kTiles) {  // DMA next tile into the other buffer, then make
        tdm_load_tile(z + (size_t)nxt * kNT * (kH * kDH) + h * kDH,
                      (unsigned)(uintptr_t)(const void*)sZf[cur ^ 1],
                      (unsigned)(kN - nxt * kNT));
        __builtin_amdgcn_s_wait_tensorcnt(1);  // ...sure current tile landed
      } else {
        __builtin_amdgcn_s_wait_tensorcnt(0);
      }
    }
    __syncthreads();  // all waves may now read sZf[cur]

    {  // convert f32 tile -> bf16 A-source (one b128 store per thread)
      int row = t >> 2, c0 = (t & 3) * 8;
      const float* src = &sZf[cur][row * kDH + c0];
      union { __bf16 hx[8]; uint4 u; } pk;
#pragma unroll
      for (int j = 0; j < 8; ++j) pk.hx[j] = (__bf16)src[j];
      *(uint4*)&sZ[row * 40 + c0] = pk.u;
    }
    __syncthreads();

    {  // GEMM1: k' and v (one 16x16 tile each per wave)
      int lane = t & 31;
      int col  = (w & 1) * 16 + (lane & 15);
      int rb   = (w >> 1) * 16 + ((lane & 16) ? 8 : 0);
      v16bf az = lds_load_a(sZ, 40, (w >> 1) * 16);
      v8f ka = {}, va = {};
      ka = wmma_bf16(az, lds_load_bfrag(sWk, w & 1), ka);
      va = wmma_bf16(az, lds_load_bfrag(sWv, w & 1), va);
      float bk = Wk_b[h * kDH + col];
      float bv = Wv_b[h * kDH + col];
#pragma unroll
      for (int i = 0; i < 8; ++i) {
        int row = rb + i;
        sK[row * 40 + col] = (__bf16)(sc * (ka[i] + bk));
        sVf[bfrag_idx64(row, col, 2)] = (__bf16)(va[i] + bv);  // B-frag, K=row
      }
    }
    __syncthreads();

#pragma unroll
    for (int i = 0; i < 2; ++i) {  // GEMM2: dd = k' @ proj^T
      int tid = w * 2 + i;
      int r = tid >> 2, c = tid & 3;
      v8f acc = {};
      acc = wmma_bf16(lds_load_a(sK, 40, r * 16), lds_load_bfrag(sPT, c), acc);
      lds_store_c(sDD, kDDld, r * 16, c * 16, acc);
    }
    if (t < kNT) {  // diag = 0.5*||k'||^2
      float d = 0.0f;
#pragma unroll
      for (int j = 0; j < 32; ++j) {
        float x = (float)sK[t * 40 + j];
        d += x * x;
      }
      sDiag[t] = 0.5f * d;
    }
    __syncthreads();

    {  // kk = ratio*(exp(dd - diag - kmax) + eps), transposed into LDS
      int  row   = t >> 2, m0 = (t & 3) * 16;
      bool valid = (n0 + row) < kN;
      float dg   = sDiag[row];
#pragma unroll
      for (int j = 0; j < 16; ++j) {
        int   m  = m0 + j;
        float kk = valid
            ? kRatio * (expf(sDD[row * kDDld + m] - dg - kmax) + kEps)
            : 0.0f;
        sKKT[m * 72 + row] = (__bf16)kk;
      }
    }
    __syncthreads();

    if (t < kM) {  // ksum accumulation
#pragma unroll
      for (int r = 0; r < kNT; ++r) ksacc += (float)sKKT[t * 72 + r];
    }
#pragma unroll
    for (int ks = 0; ks < 2; ++ks) {  // GEMM3: kvs += kk^T @ v (K = 64 rows)
      kvacc = wmma_bf16(lds_load_a(sKKT + ks * 32, 72, (w >> 1) * 16),
                        lds_load_bfrag(sVf, ks * 2 + (w & 1)), kvacc);
    }
  }

  {  // write per-slice partials (deterministic; reduced by reduce_kernel)
    int lane = t & 31;
    int col  = (w & 1) * 16 + (lane & 15);
    int rb   = (w >> 1) * 16 + ((lane & 16) ? 8 : 0);
    float* dst = pkvs + ((size_t)(slice * kH + h)) * kM * kDH;
#pragma unroll
    for (int i = 0; i < 8; ++i) dst[(rb + i) * kDH + col] = kvacc[i];
  }
  if (t < kM) pksum[(slice * kH + h) * kM + t] = ksacc;
}

// ---------------------------------------------------------------------------
__global__ __launch_bounds__(256) void reduce_kernel(
    const float* __restrict__ pkvs, const float* __restrict__ pksum,
    float* __restrict__ kvs, float* __restrict__ ksum) {
  int i = blockIdx.x * 256 + threadIdx.x;
  if (i < kH * kM * kDH) {
    float acc = 0.0f;
    for (int s = 0; s < kS; ++s) acc += pkvs[(size_t)s * kH * kM * kDH + i];
    kvs[i] = acc;
  }
  if (i < kH * kM) {
    float acc = 0.0f;
    for (int s = 0; s < kS; ++s) acc += pksum[s * kH * kM + i];
    ksum[i] = acc;
  }
}

// ---------------------------------------------------------------------------
// Pass C: qk features, numerator = qk@kvs, denominator = qk.ksum, output.
__global__ __launch_bounds__(256) void out_kernel(
    const float* __restrict__ z, const float* __restrict__ tau,
    const float* __restrict__ Wq_w, const float* __restrict__ Wq_b,
    const float* __restrict__ proj, const float* __restrict__ kvs,
    const float* __restrict__ ksum, float* __restrict__ out) {
  __shared__ __bf16 sZ[kNT * 40]  __attribute__((aligned(16)));
  __shared__ __bf16 sW[1024]      __attribute__((aligned(16)));  // B-frag
  __shared__ __bf16 sPT[2048]     __attribute__((aligned(16)));  // B-frag
  __shared__ __bf16 sQ[kNT * 40]  __attribute__((aligned(16)));
  __shared__ __bf16 sQK[kNT * 72] __attribute__((aligned(16)));
  __shared__ __bf16 sKVS[2048]    __attribute__((aligned(16)));  // B-frag K=64
  __shared__ float  sDD[kNT * kDDld];
  __shared__ float  sKS[kM];
  __shared__ float  sDiag[kNT];
  __shared__ float  sRmax[kNT];
  __shared__ float  sDen[kNT];
  __shared__ float  sRed[256];

  const int   t  = threadIdx.x;
  const int   w  = t >> 5;
  const int   h  = blockIdx.y;
  const int   n0 = blockIdx.x * kNT;
  const float sc = rsqrtf(tau[0]) * kDN;

  for (int i = t; i < 32 * 32; i += 256)
    sW[bfrag_idx64(i >> 5, i & 31, 2)] = (__bf16)Wq_w[h * 32 * 32 + i];
  for (int i = t; i < kM * 32; i += 256) {
    int m = i >> 5, e = i & 31;
    sPT[bfrag_idx64(e, m, 4)]  = (__bf16)proj[i];
    sKVS[bfrag_idx64(m, e, 2)] = (__bf16)kvs[h * kM * kDH + i];  // K = m
  }
  if (t < kM) sKS[t] = ksum[h * kM + t];
  {
    int row = t >> 2, c0 = (t & 3) * 8;
    int n = n0 + row;
    const float* src = z + (size_t)n * (kH * kDH) + h * kDH + c0;
    union { __bf16 hx[8]; uint4 u; } pk;
#pragma unroll
    for (int j = 0; j < 8; ++j)
      pk.hx[j] = (__bf16)((n < kN) ? src[j] : 0.0f);
    *(uint4*)&sZ[row * 40 + c0] = pk.u;
  }
  __syncthreads();

  {  // GEMM1: q' = sc * (z @ Wq + b)
    v8f acc = {};
    acc = wmma_bf16(lds_load_a(sZ, 40, (w >> 1) * 16),
                    lds_load_bfrag(sW, w & 1), acc);
    int lane = t & 31;
    int col  = (w & 1) * 16 + (lane & 15);
    int rb   = (w >> 1) * 16 + ((lane & 16) ? 8 : 0);
    float bias = Wq_b[h * kDH + col];
#pragma unroll
    for (int i = 0; i < 8; ++i)
      sQ[(rb + i) * 40 + col] = (__bf16)(sc * (acc[i] + bias));
  }
  __syncthreads();

#pragma unroll
  for (int i = 0; i < 2; ++i) {  // GEMM2: dd = q' @ proj^T
    int tid = w * 2 + i;
    int r = tid >> 2, c = tid & 3;
    v8f acc = {};
    acc = wmma_bf16(lds_load_a(sQ, 40, r * 16), lds_load_bfrag(sPT, c), acc);
    lds_store_c(sDD, kDDld, r * 16, c * 16, acc);
  }
  if (t < kNT) {
    float d = 0.0f;
#pragma unroll
    for (int j = 0; j < 32; ++j) {
      float x = (float)sQ[t * 40 + j];
      d += x * x;
    }
    sDiag[t] = 0.5f * d;
  }
  __syncthreads();

  {  // per-row max over m
    int   row = t >> 2, c0 = (t & 3) * 16;
    float mx  = -3.0e38f;
#pragma unroll
    for (int j = 0; j < 16; ++j) mx = fmaxf(mx, sDD[row * kDDld + c0 + j]);
    sRed[t] = mx;
  }
  __syncthreads();
  if (t < kNT)
    sRmax[t] = fmaxf(fmaxf(sRed[t * 4 + 0], sRed[t * 4 + 1]),
                     fmaxf(sRed[t * 4 + 2], sRed[t * 4 + 3]));
  __syncthreads();

  {  // qk + partial denominator
    int   row = t >> 2, m0 = (t & 3) * 16;
    float dg = sDiag[row], rm = sRmax[row], dp = 0.0f;
#pragma unroll
    for (int j = 0; j < 16; ++j) {
      int   m  = m0 + j;
      float qk = kRatio * (expf(sDD[row * kDDld + m] - dg - rm) + kEps);
      sQK[row * 72 + m] = (__bf16)qk;
      dp += qk * sKS[m];
    }
    sRed[t] = dp;
  }
  __syncthreads();
  if (t < kNT)
    sDen[t] = (sRed[t * 4 + 0] + sRed[t * 4 + 1]) +
              (sRed[t * 4 + 2] + sRed[t * 4 + 3]);
  __syncthreads();

  v8f acc = {};
#pragma unroll
  for (int ks = 0; ks < 2; ++ks)  // GEMM4: numerator = qk @ kvs (K = 64)
    acc = wmma_bf16(lds_load_a(sQK + ks * 32, 72, (w >> 1) * 16),
                    lds_load_bfrag(sKVS, ks * 2 + (w & 1)), acc);

  {  // divide by denominator, store
    int lane = t & 31;
    int col  = (w & 1) * 16 + (lane & 15);
    int rb   = (w >> 1) * 16 + ((lane & 16) ? 8 : 0);
#pragma unroll
    for (int i = 0; i < 8; ++i) {
      int row = rb + i;
      int n   = n0 + row;
      if (n < kN)
        out[(size_t)n * (kH * kDH) + h * kDH + col] = acc[i] / sDen[row];
    }
  }
}

}  // namespace

// ---------------------------------------------------------------------------
extern "C" void kernel_launch(void* const* d_in, const int* in_sizes, int n_in,
                              void* d_out, int out_size, void* d_ws,
                              size_t ws_size, hipStream_t stream) {
  (void)in_sizes; (void)n_in; (void)out_size; (void)ws_size;
  const float* z    = (const float*)d_in[0];
  const float* tau  = (const float*)d_in[1];
  const float* Wq_w = (const float*)d_in[2];
  const float* Wq_b = (const float*)d_in[3];
  const float* Wk_w = (const float*)d_in[4];
  const float* Wk_b = (const float*)d_in[5];
  const float* Wv_w = (const float*)d_in[6];
  const float* Wv_b = (const float*)d_in[7];
  const float* proj = (const float*)d_in[8];
  float* out = (float*)d_out;

  char* ws = (char*)d_ws;
  unsigned* kmax_bits = (unsigned*)ws;                       //     32 B
  float*    ksum      = (float*)(ws + 64);                   //   2 KiB
  float*    kvs       = (float*)(ws + 64 + 2048);            //  64 KiB
  float*    pkvs      = (float*)(ws + 64 + 2048 + 65536);    //   4 MiB
  float*    pksum     = (float*)(ws + 64 + 2048 + 65536 +
                                 (size_t)kS * kH * kM * kDH * sizeof(float));

  hipLaunchKernelGGL(init_kmax, dim3(1), dim3(256), 0, stream, kmax_bits);
  hipLaunchKernelGGL(kmax_kernel, dim3(kTiles, kH), dim3(256), 0, stream,
                     z, tau, Wk_w, Wk_b, proj, kmax_bits);
  hipLaunchKernelGGL(kvs_kernel, dim3(kS, kH), dim3(256), 0, stream,
                     z, tau, Wk_w, Wk_b, Wv_w, Wv_b, proj, kmax_bits,
                     pkvs, pksum);
  hipLaunchKernelGGL(reduce_kernel, dim3((kH * kM * kDH + 255) / 256),
                     dim3(256), 0, stream, pkvs, pksum, kvs, ksum);
  hipLaunchKernelGGL(out_kernel, dim3(kTiles, kH), dim3(256), 0, stream,
                     z, tau, Wq_w, Wq_b, proj, kvs, ksum, out);
}
